// DE_Func_25323127177649
// MI455X (gfx1250) — compile-verified, matrix operand order checked by reading, not run on hardware
//
#include <hip/hip_runtime.h>
#include <hip/hip_bf16.h>

typedef __attribute__((ext_vector_type(16))) __bf16 v16bf;
typedef __attribute__((ext_vector_type(8)))  __bf16 v8bf;
typedef __attribute__((ext_vector_type(8)))  float  v8f;

#define RS 72   // LDS row stride in halfs (16B-aligned rows, dodges bank conflicts)

// hardware tanh (TRANS op) on gfx1250 if the toolchain exposes it
#if __has_builtin(__builtin_amdgcn_tanh_f32)
#define FAST_TANH(x) __builtin_amdgcn_tanh_f32(x)
#else
#define FAST_TANH(x) tanhf(x)
#endif

// ---- workspace layout (offsets in bf16 elements) ----
static constexpr size_t WS_GRP  = 0;                       // 8 stacks * 32 groups * 4096
static constexpr size_t WS_V1T  = 8ull * 32 * 4096;        // 1,048,576
static constexpr size_t WS_V10  = WS_V1T + 4096;
static constexpr size_t WS_V2   = WS_V10 + 4096;
static constexpr size_t WS_V3   = WS_V2  + 4096;
static constexpr size_t WS_V4   = WS_V3  + 4096;
static constexpr size_t WS_FT   = WS_V4  + 2048;           // 1,067,008 (16B aligned)
static constexpr size_t WS_F0   = WS_FT + 8192ull * 4096;  // [B,64,64] bf16
static constexpr size_t WS_VOUT = WS_F0 + 8192ull * 4096;  // [B,32,64] bf16

__device__ __forceinline__ float actf(float x, int kind) {
  if (kind == 0) return FAST_TANH(x);
  if (kind == 1) return x > 0.f ? x : (__expf(x) - 1.f);
  return x;
}

// A-fragment (16x32 bf16) from LDS per ISA layout:
// lanes 0-15: VGPR0-3 = K0..7, VGPR4-7 = K16..23 ; lanes 16-31: K8..15 / K24..31
__device__ __forceinline__ v16bf lds_afrag(const __bf16* a, int lane, int k0) {
  const int m = lane & 15;
  const int hi8 = ((lane >> 4) & 1) * 8;
  v8bf lo = *(const v8bf*)(a + m * RS + k0 + hi8);
  v8bf hi = *(const v8bf*)(a + m * RS + k0 + 16 + hi8);
  v16bf r;
#pragma unroll
  for (int j = 0; j < 8; ++j) { r[j] = lo[j]; r[8 + j] = hi[j]; }
  return r;
}

// B-fragment: pre-swizzled by kPrep -> contiguous 32B per lane
__device__ __forceinline__ v16bf glb_bfrag(const __bf16* wf, int lane) {
  return *(const v16bf*)(wf + lane * 16);
}

// one 16x16 output tile (ntile nt), K=64: 2 x v_wmma_f32_16x16x32_bf16
__device__ __forceinline__ v8f wmma_k64(const __bf16* aIn, const __bf16* wf,
                                        const float* bias, int lane, int nt) {
  float bv = bias[nt * 16 + (lane & 15)];
  v8f c;
#pragma unroll
  for (int r = 0; r < 8; ++r) c[r] = bv;
#pragma unroll
  for (int kt = 0; kt < 2; ++kt) {
    v16bf a = lds_afrag(aIn, lane, kt * 32);
    v16bf b = glb_bfrag(wf + (size_t)(kt * 4 + nt) * 512, lane);
    c = __builtin_amdgcn_wmma_f32_16x16x32_bf16(false, a, false, b, (short)0, c, false, false);
  }
  return c;
}

// quad 16x16 tiles (four m-tiles of a 64-row A), shared B fragment:
// 4 independent accumulator chains -> covers the WMMA D->C / WMMA->VALU hazard window
__device__ __forceinline__ void wmma_k64x4(const __bf16* aIn, const __bf16* wf,
                                           const float* bias, int lane, int nt, v8f c[4]) {
  float bv = bias[nt * 16 + (lane & 15)];
#pragma unroll
  for (int t = 0; t < 4; ++t)
#pragma unroll
    for (int r = 0; r < 8; ++r) c[t][r] = bv;
#pragma unroll
  for (int kt = 0; kt < 2; ++kt) {
    v16bf b = glb_bfrag(wf + (size_t)(kt * 4 + nt) * 512, lane);
#pragma unroll
    for (int t = 0; t < 4; ++t) {
      v16bf a = lds_afrag(aIn + t * 16 * RS, lane, kt * 32);
      c[t] = __builtin_amdgcn_wmma_f32_16x16x32_bf16(false, a, false, b, (short)0, c[t], false, false);
    }
  }
}

// D layout: lane<16 -> m=r, lane>=16 -> m=r+8 ; n = lane&15 (+16*nt)
__device__ __forceinline__ void store_lds(__bf16* dst, int lane, int nt, v8f c, int kind) {
  const int n = nt * 16 + (lane & 15);
  const int mo = ((lane >> 4) & 1) * 8;
#pragma unroll
  for (int r = 0; r < 8; ++r) dst[(mo + r) * RS + n] = (__bf16)actf(c[r], kind);
}

__device__ __forceinline__ void store_glb_bf16(__bf16* dst, int rstride, int lane, int nt, v8f c) {
  const int n = nt * 16 + (lane & 15);
  const int mo = ((lane >> 4) & 1) * 8;
#pragma unroll
  for (int r = 0; r < 8; ++r) dst[(size_t)(mo + r) * rstride + n] = (__bf16)c[r];
}

// quad-tile helpers over a 64-row activation buffer
__device__ __forceinline__ void layer64_lds(const __bf16* aIn, __bf16* aOut, const __bf16* wf,
                                            const float* bias, int lane, int nt, int kind) {
  v8f c[4];
  wmma_k64x4(aIn, wf, bias, lane, nt, c);
#pragma unroll
  for (int t = 0; t < 4; ++t) store_lds(aOut + t * 16 * RS, lane, nt, c[t], kind);
}

__device__ __forceinline__ void layer64_glb(const __bf16* aIn, __bf16* gout, int rstride,
                                            const __bf16* wf, const float* bias, int lane, int nt) {
  v8f c[4];
  wmma_k64x4(aIn, wf, bias, lane, nt, c);
#pragma unroll
  for (int t = 0; t < 4; ++t)
    store_glb_bf16(gout + (size_t)(t * 16) * rstride, rstride, lane, nt, c[t]);
}

// ---------------- kernel 0: weight prep (f32 -> bf16, WMMA-swizzled) ----------------
__global__ __launch_bounds__(256) void kPrep(
    const float* __restrict__ xencW2, const float* __restrict__ zencW2,
    const float* __restrict__ xextW1, const float* __restrict__ xextW2,
    const float* __restrict__ zextW1, const float* __restrict__ zextW2,
    const float* __restrict__ xdotW1, const float* __restrict__ xdotW2,
    const float* __restrict__ V1, const float* __restrict__ V2,
    const float* __restrict__ V3, const float* __restrict__ V4,
    __bf16* __restrict__ ws) {
  const int blk = blockIdx.x, t = threadIdx.x;
  if (blk < 256) {                      // 8 group stacks x 32 groups, each [64K x 64N]
    const float* srcs[8] = {xencW2, zencW2, xextW1, xextW2, zextW1, zextW2, xdotW1, xdotW2};
    const int s = blk >> 5, g = blk & 31;
    const float* src = srcs[s] + (size_t)g * 4096;
    __bf16* dst = ws + WS_GRP + (size_t)blk * 4096;
    for (int q = t; q < 4096; q += 256) {
      int f = q >> 9, L = (q >> 4) & 31, jj = q & 15, kt = f >> 2, nt = f & 3;
      int k = kt * 32 + ((jj & 8) ? 16 : 0) + (((L >> 4) & 1) ? 8 : 0) + (jj & 7);
      int n = nt * 16 + (L & 15);
      dst[q] = (__bf16)src[k * 64 + n];
    }
  } else if (blk == 256 || blk == 257) { // V1t = V1a+V1c ; V10 = V1b-V1c (folds the concat/diff)
    const bool isT = (blk == 256);
    __bf16* dst = ws + (isT ? WS_V1T : WS_V10);
    for (int q = t; q < 4096; q += 256) {
      int f = q >> 9, L = (q >> 4) & 31, jj = q & 15, kt = f >> 2, nt = f & 3;
      int k = kt * 32 + ((jj & 8) ? 16 : 0) + (((L >> 4) & 1) ? 8 : 0) + (jj & 7);
      int n = nt * 16 + (L & 15);
      float v = isT ? (V1[k * 64 + n] + V1[(128 + k) * 64 + n])
                    : (V1[(64 + k) * 64 + n] - V1[(128 + k) * 64 + n]);
      dst[q] = (__bf16)v;
    }
  } else if (blk == 258 || blk == 259) { // V2 / V3 [64x64]
    const float* src = (blk == 258) ? V2 : V3;
    __bf16* dst = ws + ((blk == 258) ? WS_V2 : WS_V3);
    for (int q = t; q < 4096; q += 256) {
      int f = q >> 9, L = (q >> 4) & 31, jj = q & 15, kt = f >> 2, nt = f & 3;
      int k = kt * 32 + ((jj & 8) ? 16 : 0) + (((L >> 4) & 1) ? 8 : 0) + (jj & 7);
      int n = nt * 16 + (L & 15);
      dst[q] = (__bf16)src[k * 64 + n];
    }
  } else {                              // V4 [64K x 32N] -> 2 ntiles
    __bf16* dst = ws + WS_V4;
    for (int q = t; q < 2048; q += 256) {
      int f = q >> 9, L = (q >> 4) & 31, jj = q & 15, kt = f >> 1, nt = f & 1;
      int k = kt * 32 + ((jj & 8) ? 16 : 0) + (((L >> 4) & 1) ? 8 : 0) + (jj & 7);
      int n = nt * 16 + (L & 15);
      dst[q] = (__bf16)V4[k * 32 + n];
    }
  }
}

// ---------------- kernel 1: grouped encoder/extractor features ----------------
// grid (64 groups, 128 b-tiles of 64 rows), 4 waves; writes f_t, f_0 bf16 [B,64,64]
__global__ __launch_bounds__(128) void kFeat(
    const float* __restrict__ Xht, const float* __restrict__ zt,
    const float* __restrict__ x0,  const float* __restrict__ z0,
    const float* __restrict__ xencW1, const float* __restrict__ xencB1, const float* __restrict__ xencB2,
    const float* __restrict__ zencW1, const float* __restrict__ zencB1, const float* __restrict__ zencB2,
    const float* __restrict__ xextB1, const float* __restrict__ xextB2,
    const float* __restrict__ zextB1, const float* __restrict__ zextB2,
    __bf16* __restrict__ ws) {
  const int g = blockIdx.x, b0 = blockIdx.y * 64;
  const int tid = threadIdx.x, lane = tid & 31, wid = tid >> 5;
  __shared__ __bf16 sA[64 * RS], sB[64 * RS];
  __bf16* ft = ws + WS_FT + ((size_t)b0 * 64 + g) * 64;
  __bf16* f0 = ws + WS_F0 + ((size_t)b0 * 64 + g) * 64;
  if (g < 32) {
    const __bf16* extW1 = ws + WS_GRP + (size_t)(2 * 32 + g) * 4096;
    const __bf16* extW2 = ws + WS_GRP + (size_t)(3 * 32 + g) * 4096;
    const __bf16* encW2 = ws + WS_GRP + (size_t)(0 * 32 + g) * 4096;
    // path t: f_t = elu(Xht*W1+b1)*W2 + b2
    for (int t = tid; t < 4096; t += 128) {
      int bb = t >> 6, h = t & 63;
      sA[bb * RS + h] = (__bf16)Xht[((size_t)(b0 + bb) * 32 + g) * 64 + h];
    }
    __syncthreads();
    layer64_lds(sA, sB, extW1, xextB1 + g * 64, lane, wid, 1);
    __syncthreads();
    layer64_glb(sB, ft, 4096, extW2, xextB2 + g * 64, lane, wid);
    // path 0: Xh0 = tanh(x0*w+b)*W2+b2 ; f_0 = elu(Xh0*W1+b1)*W2+b2
    for (int t = tid; t < 4096; t += 128) {
      int bb = t >> 6, h = t & 63;
      sA[bb * RS + h] =
          (__bf16)FAST_TANH(x0[(size_t)(b0 + bb) * 32 + g] * xencW1[g * 64 + h] + xencB1[g * 64 + h]);
    }
    __syncthreads();
    layer64_lds(sA, sB, encW2, xencB2 + g * 64, lane, wid, 2);
    __syncthreads();
    layer64_lds(sB, sA, extW1, xextB1 + g * 64, lane, wid, 1);
    __syncthreads();
    layer64_glb(sA, f0, 4096, extW2, xextB2 + g * 64, lane, wid);
  } else {
    const int j = g - 32;
    const __bf16* extW1 = ws + WS_GRP + (size_t)(4 * 32 + j) * 4096;
    const __bf16* extW2 = ws + WS_GRP + (size_t)(5 * 32 + j) * 4096;
    const __bf16* encW2 = ws + WS_GRP + (size_t)(1 * 32 + j) * 4096;
    // path t (zt): tanh ew -> linear GEMM -> elu GEMM -> linear GEMM
    for (int t = tid; t < 4096; t += 128) {
      int bb = t >> 6, h = t & 63;
      sA[bb * RS + h] =
          (__bf16)FAST_TANH(zt[(size_t)(b0 + bb) * 32 + j] * zencW1[j * 64 + h] + zencB1[j * 64 + h]);
    }
    __syncthreads();
    layer64_lds(sA, sB, encW2, zencB2 + j * 64, lane, wid, 2);
    __syncthreads();
    layer64_lds(sB, sA, extW1, zextB1 + j * 64, lane, wid, 1);
    __syncthreads();
    layer64_glb(sA, ft, 4096, extW2, zextB2 + j * 64, lane, wid);
    // path 0 (z0)
    for (int t = tid; t < 4096; t += 128) {
      int bb = t >> 6, h = t & 63;
      sB[bb * RS + h] =
          (__bf16)FAST_TANH(z0[(size_t)(b0 + bb) * 32 + j] * zencW1[j * 64 + h] + zencB1[j * 64 + h]);
    }
    __syncthreads();
    layer64_lds(sB, sA, encW2, zencB2 + j * 64, lane, wid, 2);
    __syncthreads();
    layer64_lds(sA, sB, extW1, zextB1 + j * 64, lane, wid, 1);
    __syncthreads();
    layer64_glb(sB, f0, 4096, extW2, zextB2 + j * 64, lane, wid);
  }
}

// ---------------- kernel 2: shared V-MLP over rows (b,h) ----------------
// one block per batch row b; 4 waves each own a 16-row h-tile (4 independent nt chains/wave)
__global__ __launch_bounds__(128) void kVmlp(
    __bf16* __restrict__ ws,
    const float* __restrict__ vb1, const float* __restrict__ vb2,
    const float* __restrict__ vb3, const float* __restrict__ vb4) {
  const int b = blockIdx.x, tid = threadIdx.x, lane = tid & 31, rt = tid >> 5;
  __shared__ __bf16 ftT[64 * RS], f0T[64 * RS], acta[64 * RS], actb[64 * RS], outT[32 * RS];
  const __bf16* ft = ws + WS_FT + (size_t)b * 4096;
  const __bf16* f0 = ws + WS_F0 + (size_t)b * 4096;
  // coalesced load + LDS transpose: A[h][g] = f[b,g,h]
  for (int idx = tid; idx < 512; idx += 128) {
    int gg = idx >> 3, h0 = (idx & 7) * 8;
    v8bf x = *(const v8bf*)(ft + gg * 64 + h0);
    v8bf y = *(const v8bf*)(f0 + gg * 64 + h0);
#pragma unroll
    for (int r = 0; r < 8; ++r) { ftT[(h0 + r) * RS + gg] = x[r]; f0T[(h0 + r) * RS + gg] = y[r]; }
  }
  __syncthreads();
  const __bf16* V1ts = ws + WS_V1T;
  const __bf16* V10s = ws + WS_V10;
  const __bf16* V2s  = ws + WS_V2;
  const __bf16* V3s  = ws + WS_V3;
  const __bf16* V4s  = ws + WS_V4;
  const __bf16* aF = ftT + rt * 16 * RS;
  const __bf16* aZ = f0T + rt * 16 * RS;
  __bf16* o1 = acta + rt * 16 * RS;
  __bf16* o2 = actb + rt * 16 * RS;
  // layer 1: K=128 (ft part x V1t + f0 part x V10), elu
#pragma unroll
  for (int nt = 0; nt < 4; ++nt) {
    float bv = vb1[nt * 16 + (lane & 15)];
    v8f c;
#pragma unroll
    for (int r = 0; r < 8; ++r) c[r] = bv;
#pragma unroll
    for (int kt = 0; kt < 2; ++kt) {
      v16bf a = lds_afrag(aF, lane, kt * 32);
      v16bf w = glb_bfrag(V1ts + (size_t)(kt * 4 + nt) * 512, lane);
      c = __builtin_amdgcn_wmma_f32_16x16x32_bf16(false, a, false, w, (short)0, c, false, false);
    }
#pragma unroll
    for (int kt = 0; kt < 2; ++kt) {
      v16bf a = lds_afrag(aZ, lane, kt * 32);
      v16bf w = glb_bfrag(V10s + (size_t)(kt * 4 + nt) * 512, lane);
      c = __builtin_amdgcn_wmma_f32_16x16x32_bf16(false, a, false, w, (short)0, c, false, false);
    }
    store_lds(o1, lane, nt, c, 1);
  }
  // layers 2,3: K=64, elu (wave-private rows: no block barrier needed)
#pragma unroll
  for (int nt = 0; nt < 4; ++nt) { v8f c = wmma_k64(o1, V2s, vb2, lane, nt); store_lds(o2, lane, nt, c, 1); }
#pragma unroll
  for (int nt = 0; nt < 4; ++nt) { v8f c = wmma_k64(o2, V3s, vb3, lane, nt); store_lds(o1, lane, nt, c, 1); }
  // layer 4: K=64 -> N=32, linear; store transposed into outT[i][h]
#pragma unroll
  for (int nt = 0; nt < 2; ++nt) {
    float bv = vb4[nt * 16 + (lane & 15)];
    v8f c;
#pragma unroll
    for (int r = 0; r < 8; ++r) c[r] = bv;
#pragma unroll
    for (int kt = 0; kt < 2; ++kt) {
      v16bf a = lds_afrag(o1, lane, kt * 32);
      v16bf w = glb_bfrag(V4s + (size_t)(kt * 2 + nt) * 512, lane);
      c = __builtin_amdgcn_wmma_f32_16x16x32_bf16(false, a, false, w, (short)0, c, false, false);
    }
    const int n = nt * 16 + (lane & 15);
    const int mo = ((lane >> 4) & 1) * 8;
#pragma unroll
    for (int r = 0; r < 8; ++r) outT[n * RS + rt * 16 + mo + r] = (__bf16)c[r];
  }
  __syncthreads();
  // write vout[b, i, h] (already the [B,XD,H] layout the final kernel wants)
  __bf16* vout = ws + WS_VOUT + (size_t)b * 2048;
  for (int idx = tid; idx < 256; idx += 128) {
    int i = idx >> 3, h0 = (idx & 7) * 8;
    *(v8bf*)(vout + i * 64 + h0) = *(const v8bf*)(outT + i * RS + h0);
  }
}

// ---------------- kernel 3: final grouped xdot MLP -> f32 output ----------------
// grid (32 groups, 128 b-tiles of 64 rows)
__global__ __launch_bounds__(128) void kDot(
    const __bf16* __restrict__ ws,
    const float* __restrict__ xdotB1, const float* __restrict__ xdotB2,
    float* __restrict__ out) {
  const int i = blockIdx.x, b0 = blockIdx.y * 64;
  const int tid = threadIdx.x, lane = tid & 31, wid = tid >> 5;
  __shared__ __bf16 sA[64 * RS], sB[64 * RS];
  const __bf16* vout = ws + WS_VOUT;
  for (int t = tid; t < 4096; t += 128) {
    int bb = t >> 6, h = t & 63;
    sA[bb * RS + h] = vout[((size_t)(b0 + bb) * 32 + i) * 64 + h];
  }
  __syncthreads();
  const __bf16* w1s = ws + WS_GRP + (size_t)(6 * 32 + i) * 4096;
  const __bf16* w2s = ws + WS_GRP + (size_t)(7 * 32 + i) * 4096;
  layer64_lds(sA, sB, w1s, xdotB1 + i * 64, lane, wid, 1);
  __syncthreads();
  { v8f c[4];
    wmma_k64x4(sB, w2s, xdotB2 + i * 64, lane, wid, c);
    const int n = wid * 16 + (lane & 15);
    const int mo = ((lane >> 4) & 1) * 8;
#pragma unroll
    for (int t = 0; t < 4; ++t)
#pragma unroll
      for (int r = 0; r < 8; ++r)
        out[((size_t)(b0 + t * 16 + mo + r) * 32 + i) * 64 + n] = c[t][r];
  }
}

extern "C" void kernel_launch(void* const* d_in, const int* in_sizes, int n_in,
                              void* d_out, int out_size, void* d_ws, size_t ws_size,
                              hipStream_t stream) {
  (void)in_sizes; (void)n_in; (void)out_size; (void)ws_size;
  const float* Xht    = (const float*)d_in[1];
  const float* zt     = (const float*)d_in[2];
  const float* x0     = (const float*)d_in[3];
  const float* z0     = (const float*)d_in[4];
  const float* xencW1 = (const float*)d_in[5];
  const float* xencB1 = (const float*)d_in[6];
  const float* xencW2 = (const float*)d_in[7];
  const float* xencB2 = (const float*)d_in[8];
  const float* zencW1 = (const float*)d_in[9];
  const float* zencB1 = (const float*)d_in[10];
  const float* zencW2 = (const float*)d_in[11];
  const float* zencB2 = (const float*)d_in[12];
  const float* xextW1 = (const float*)d_in[13];
  const float* xextB1 = (const float*)d_in[14];
  const float* xextW2 = (const float*)d_in[15];
  const float* xextB2 = (const float*)d_in[16];
  const float* zextW1 = (const float*)d_in[17];
  const float* zextB1 = (const float*)d_in[18];
  const float* zextW2 = (const float*)d_in[19];
  const float* zextB2 = (const float*)d_in[20];
  const float* xdotW1 = (const float*)d_in[21];
  const float* xdotB1 = (const float*)d_in[22];
  const float* xdotW2 = (const float*)d_in[23];
  const float* xdotB2 = (const float*)d_in[24];
  const float* V1     = (const float*)d_in[25];
  const float* vb1    = (const float*)d_in[26];
  const float* V2     = (const float*)d_in[27];
  const float* vb2    = (const float*)d_in[28];
  const float* V3     = (const float*)d_in[29];
  const float* vb3    = (const float*)d_in[30];
  const float* V4     = (const float*)d_in[31];
  const float* vb4    = (const float*)d_in[32];
  __bf16* ws = (__bf16*)d_ws;

  kPrep<<<261, 256, 0, stream>>>(xencW2, zencW2, xextW1, xextW2, zextW1, zextW2,
                                 xdotW1, xdotW2, V1, V2, V3, V4, ws);
  kFeat<<<dim3(64, 128), 128, 0, stream>>>(Xht, zt, x0, z0,
                                           xencW1, xencB1, xencB2,
                                           zencW1, zencB1, zencB2,
                                           xextB1, xextB2, zextB1, zextB2, ws);
  kVmlp<<<8192, 128, 0, stream>>>(ws, vb1, vb2, vb3, vb4);
  kDot<<<dim3(32, 128), 128, 0, stream>>>(ws, xdotB1, xdotB2, (float*)d_out);
}
